// ResidualVectorQuantization_89747636617345
// MI455X (gfx1250) — matrix operand order repaired and checked by reading, not compiled, not run on hardware
//
#include <hip/hip_runtime.h>
#include <stdint.h>

typedef __attribute__((ext_vector_type(16))) __bf16 v16bf;
typedef __attribute__((ext_vector_type(8)))  float  v8f;
typedef __attribute__((ext_vector_type(4)))  int    v4i;

typedef __attribute__((address_space(1))) v4i gas_v4i;   // global
typedef __attribute__((address_space(3))) v4i las_v4i;   // LDS

#define N_ROWS 96000
#define DIM    512
#define NCODES 1024
#define NQUANT 8

#define MT  32            // rows per workgroup
#define KT  128           // codes per K-tile (2 16-wide subtiles per wave)
#define DC  32            // depth chunk (bf16 WMMA K)
#define AST 520           // A LDS row stride (bf16), padded for bank spread
#define BST 40            // B LDS row stride (bf16), padded for bank spread
#define NDC (DIM / DC)    // 16 depth chunks
#define NKT (NCODES / KT) // 8 K-tiles

#if __has_builtin(__builtin_amdgcn_global_load_async_to_lds_b128)
#  define USE_ASYNC_LDS 1
#else
#  define USE_ASYNC_LDS 0
#endif

#if __has_builtin(__builtin_amdgcn_s_wait_asynccnt)
#  define ASYNC_WAIT0() __builtin_amdgcn_s_wait_asynccnt(0)
#else
#  define ASYNC_WAIT0() asm volatile("s_wait_asynccnt 0x0" ::: "memory")
#endif

// async global->LDS 16B copy (per-lane addresses)
__device__ __forceinline__ void async_b128(const unsigned short* g,
                                           unsigned short* l) {
#if USE_ASYNC_LDS
    __builtin_amdgcn_global_load_async_to_lds_b128((gas_v4i*)g, (las_v4i*)l,
                                                   0, 0);
#else
    *(uint4*)l = *(const uint4*)g;
#endif
}

// round-to-nearest-even fp32 -> bf16
__device__ __forceinline__ unsigned short f2bf(float f) {
    unsigned int u = __float_as_uint(f);
    unsigned int r = (u + 0x7FFFu + ((u >> 16) & 1u)) >> 16;
    return (unsigned short)r;
}

__device__ __forceinline__ unsigned long long kmin(unsigned long long a,
                                                   unsigned long long b) {
    return a < b ? a : b;
}

// ---------------------------------------------------------------------------
// Prep: convert one codebook layer to bf16 and compute |e_k|^2 in fp32.
// ---------------------------------------------------------------------------
__global__ void rvq_prep(const float* __restrict__ cb,
                         unsigned short* __restrict__ cbb,
                         float* __restrict__ c2) {
    __shared__ float red[256];
    const int k = blockIdx.x, t = threadIdx.x;
    const float* row = cb + (size_t)k * DIM;
    const float f0 = row[t];
    const float f1 = row[t + 256];
    cbb[(size_t)k * DIM + t]       = f2bf(f0);
    cbb[(size_t)k * DIM + t + 256] = f2bf(f1);
    red[t] = f0 * f0 + f1 * f1;
    __syncthreads();
    for (int s = 128; s > 0; s >>= 1) {
        if (t < s) red[t] += red[t + s];
        __syncthreads();
    }
    if (t == 0) c2[k] = red[0];
}

// ---------------------------------------------------------------------------
// Fused GEMM + argmin. 256 threads = 8 wave32s. M-tile 32, K-tile 128.
// Wave w: rows (w&1)*16.., cols (w>>1)*16 and (w>>1)*16+64 (2 tiles, shared
// A fragment). B chunks streamed via async global->LDS (double buffered,
// one barrier per chunk). argmin via ordered-key ds_min_u64 in LDS.
// ---------------------------------------------------------------------------
__global__ void __launch_bounds__(256)
rvq_search(const float* __restrict__ res, const unsigned short* __restrict__ cbb,
           const float* __restrict__ c2, unsigned int* __restrict__ bestIdx,
           float* __restrict__ outIdx, int q) {
    __shared__ unsigned short lA[MT * AST];          // 33280 B
    __shared__ unsigned short lB[2][KT * BST];       // 20480 B
    __shared__ unsigned long long best[MT];          //   256 B

    const int t    = threadIdx.x;
    const int lane = t & 31;
    const int wave = t >> 5;
    const int l16  = lane & 15;
    const int h    = lane >> 4;
    const int mBase  = (wave & 1) << 4;    // 0 or 16
    const int nBase0 = (wave >> 1) << 4;   // 0,16,32,48 (second tile: +64)
    const size_t rowBase = (size_t)blockIdx.x * MT;

    // Stage A tile: 32 rows x 512 fp32 -> bf16 LDS
    for (int i = 0; i < 16; ++i) {
        const int idx = t + i * 256;
        const int r   = idx >> 7;
        const int c4  = idx & 127;
        const float4 v = *(const float4*)(res + (rowBase + r) * DIM + c4 * 4);
        unsigned long long p = (unsigned long long)f2bf(v.x)
                             | ((unsigned long long)f2bf(v.y) << 16)
                             | ((unsigned long long)f2bf(v.z) << 32)
                             | ((unsigned long long)f2bf(v.w) << 48);
        *(unsigned long long*)&lA[r * AST + c4 * 4] = p;
    }
    if (t < MT) best[t] = ~0ull;

    // B staging: 128 codes x 32 bf16 = 512 x 16B; thread covers slots t, t+256
    const int c0 = t >> 2;     // code 0..63 (slot1 -> code c0+64)
    const int s0 = t & 3;      // 8-bf16 segment

    for (int kt = 0; kt < NKT; ++kt) {
        const int ktBase = kt * KT;
        v8f acc0 = {0.f, 0.f, 0.f, 0.f, 0.f, 0.f, 0.f, 0.f};
        v8f acc1 = {0.f, 0.f, 0.f, 0.f, 0.f, 0.f, 0.f, 0.f};

        // stage chunk 0 into buffer 0
        {
            const unsigned short* g0 =
                cbb + (size_t)(ktBase + c0) * DIM + s0 * 8;
            async_b128(g0,            &lB[0][c0 * BST + s0 * 8]);
            async_b128(g0 + 64 * DIM, &lB[0][(c0 + 64) * BST + s0 * 8]);
        }

        #pragma unroll
        for (int dc = 0; dc < NDC; ++dc) {
            const int buf = dc & 1;
#if USE_ASYNC_LDS
            ASYNC_WAIT0();              // own chunk-dc async writes done
#endif
            __syncthreads();            // chunk dc visible; compute(dc-1) done

            if (dc + 1 < NDC) {         // stage chunk dc+1 into other buffer
                const unsigned short* g0 = cbb +
                    (size_t)(ktBase + c0) * DIM + (dc + 1) * DC + s0 * 8;
                async_b128(g0,            &lB[buf ^ 1][c0 * BST + s0 * 8]);
                async_b128(g0 + 64 * DIM, &lB[buf ^ 1][(c0 + 64) * BST + s0 * 8]);
            }

            union { v16bf v; uint4 u4[2]; } a, b0, b1;
            // A frag (16x32 bf16): per-lane contiguous K 16h..16h+15
            const int ar = (mBase + l16) * AST + dc * DC + 16 * h;
            a.u4[0] = *(const uint4*)&lA[ar];
            a.u4[1] = *(const uint4*)&lA[ar + 8];
            // B frags (32x16 bf16): per-lane contiguous K 16h..16h+15
            const int br0 = (nBase0 + l16) * BST + 16 * h;
            b0.u4[0] = *(const uint4*)&lB[buf][br0];
            b0.u4[1] = *(const uint4*)&lB[buf][br0 + 8];
            const int br1 = (nBase0 + 64 + l16) * BST + 16 * h;
            b1.u4[0] = *(const uint4*)&lB[buf][br1];
            b1.u4[1] = *(const uint4*)&lB[buf][br1 + 8];

            acc0 = __builtin_amdgcn_wmma_f32_16x16x32_bf16(
                       false, a.v, false, b0.v, (short)0, acc0, false, false);
            acc1 = __builtin_amdgcn_wmma_f32_16x16x32_bf16(
                       false, a.v, false, b1.v, (short)0, acc1, false, false);
        }

        // dist[k] = |e_k|^2 - 2*(r.e_k); |r|^2 constant across k -> dropped
        #pragma unroll
        for (int tile = 0; tile < 2; ++tile) {
            const int kg = ktBase + nBase0 + tile * 64 + l16;
            const float c2v = c2[kg];
            const v8f& acc = tile ? acc1 : acc0;
            #pragma unroll
            for (int v = 0; v < 8; ++v) {
                const float dist = c2v - 2.0f * acc[v];
                unsigned int u = __float_as_uint(dist);
                u = (u & 0x80000000u) ? ~u : (u | 0x80000000u);
                unsigned long long key =
                    ((unsigned long long)u << 32) | (unsigned int)kg;
                key = kmin(key, __shfl_xor(key, 1, 16));
                key = kmin(key, __shfl_xor(key, 2, 16));
                key = kmin(key, __shfl_xor(key, 4, 16));
                key = kmin(key, __shfl_xor(key, 8, 16));
                if (l16 == 0) atomicMin(&best[mBase + v + 8 * h], key);
            }
        }
    }
    __syncthreads();
    if (t < MT) {
        const unsigned int idx = (unsigned int)(best[t] & 0xFFFFFFFFull);
        bestIdx[rowBase + t] = idx;
        outIdx[(size_t)q * N_ROWS + rowBase + t] = (float)idx;
    }
}

// ---------------------------------------------------------------------------
// Residual update (exact fp32): r -= e[best]; last layer also emits x - r.
// ---------------------------------------------------------------------------
__global__ void rvq_update(float* __restrict__ res, const float* __restrict__ cb,
                           const unsigned int* __restrict__ bestIdx,
                           const float* __restrict__ x, float* __restrict__ outQ,
                           int last) {
    const size_t row = blockIdx.x;
    const int t = threadIdx.x;
    const unsigned int k = bestIdx[row];
    float4* r4 = (float4*)(res + row * DIM) + t;
    const float4 c = *((const float4*)(cb + (size_t)k * DIM) + t);
    float4 r = *r4;
    r.x -= c.x; r.y -= c.y; r.z -= c.z; r.w -= c.w;
    *r4 = r;
    if (last) {
        const float4 xv = *((const float4*)(x + row * DIM) + t);
        float4 o;
        o.x = xv.x - r.x; o.y = xv.y - r.y; o.z = xv.z - r.z; o.w = xv.w - r.w;
        *((float4*)(outQ + row * DIM) + t) = o;
    }
}

// ---------------------------------------------------------------------------
extern "C" void kernel_launch(void* const* d_in, const int* in_sizes, int n_in,
                              void* d_out, int out_size, void* d_ws, size_t ws_size,
                              hipStream_t stream) {
    const float* x         = (const float*)d_in[0];   // [16,6000,512] f32
    const float* codebooks = (const float*)d_in[1];   // [8,1024,512]  f32

    float* out    = (float*)d_out;
    float* outIdx = out;                               // [8, 96000] (as f32)
    float* outQ   = out + (size_t)NQUANT * N_ROWS;     // [96000, 512]

    char* ws = (char*)d_ws;
    float*          res     = (float*)ws;                            // 196,608,000 B
    unsigned short* cbb     = (unsigned short*)(ws + 196608000);     //   1,048,576 B
    float*          c2      = (float*)(ws + 197656576);              //       4,096 B
    unsigned int*   bestIdx = (unsigned int*)(ws + 197660672);       //     384,000 B

    // residual := x
    (void)hipMemcpyAsync(res, x, (size_t)N_ROWS * DIM * sizeof(float),
                         hipMemcpyDeviceToDevice, stream);

    for (int q = 0; q < NQUANT; ++q) {
        const float* cbq = codebooks + (size_t)q * NCODES * DIM;
        rvq_prep  <<<NCODES,       256, 0, stream>>>(cbq, cbb, c2);
        rvq_search<<<N_ROWS / MT,  256, 0, stream>>>(res, cbb, c2, bestIdx,
                                                     outIdx, q);
        rvq_update<<<N_ROWS,       128, 0, stream>>>(res, cbq, bestIdx, x, outQ,
                                                     q == NQUANT - 1);
    }
}
